// VectorQuantizer_72988674228437
// MI455X (gfx1250) — matrix operand order repaired and checked by reading, not compiled, not run on hardware
//
#include <hip/hip_runtime.h>
#include <hip/hip_bf16.h>
#include <stdint.h>

typedef __bf16 bf16_t;
typedef __attribute__((ext_vector_type(16))) __bf16 v16bf;
typedef __attribute__((ext_vector_type(8)))  float  v8f;

#define K_CODES   2048
#define C_DIM     64
#define IMG_HW    (64 * 64)          // 4096
#define IMG_CHW   (C_DIM * IMG_HW)   // 262144
#define N_PIX     (32 * IMG_HW)      // 131072
#define N_CHUNKS  (K_CODES / 16)     // 128
#define WAVES_PER_BLOCK 8
#define PIX_PER_WAVE    32           // two 16-row WMMA tiles per wave
#define PIX_PER_BLOCK   (WAVES_PER_BLOCK * PIX_PER_WAVE)   // 256
#define N_BLOCKS        (N_PIX / PIX_PER_BLOCK)            // 512
#define N_WAVES         (N_PIX / PIX_PER_WAVE)             // 4096
#define CB_DWORDS       (N_CHUNKS * 2 * 32 * 8)            // 65536 dwords = 256 KB

__device__ __forceinline__ uint32_t pack_bf16(float a, float b) {
    bf16_t x = (bf16_t)a, y = (bf16_t)b;
    uint16_t ux = __builtin_bit_cast(uint16_t, x);
    uint16_t uy = __builtin_bit_cast(uint16_t, y);
    return (uint32_t)ux | ((uint32_t)uy << 16);
}

// ---- Prep kernel 1: per-code squared norms ‖e_k‖^2 -------------------------
__global__ __launch_bounds__(256) void vq_norms(const float* __restrict__ E,
                                                float* __restrict__ norms) {
    int k = blockIdx.x * 256 + threadIdx.x;
    if (k < K_CODES) {
        float s = 0.f;
        #pragma unroll 8
        for (int c = 0; c < C_DIM; ++c) { float e = E[k * C_DIM + c]; s += e * e; }
        norms[k] = s;
    }
}

// ---- Prep kernel 2: codebook -> bf16, scaled by -2, pre-swizzled into the
// WMMA B-matrix layout in contiguous k order, matching A (32x16 bf16 tiles):
//   dword index = chunk*512 + kstep*256 + lane*8 + v
//   lane = (code column n = lane&15, khalf = lane>>4), VGPR v holds K pair
//   K_local = kstep*32 + khalf*16 + 2v + {0,1}  (channel index)
__global__ __launch_bounds__(256) void vq_swizzle_cb(const float* __restrict__ E,
                                                     uint32_t* __restrict__ cbw) {
    int id    = blockIdx.x * 256 + threadIdx.x;   // 0 .. 65535
    int v     = id & 7;
    int lane  = (id >> 3) & 31;
    int s     = (id >> 8) & 1;
    int chunk = id >> 9;
    int row   = lane & 15;
    int khalf = lane >> 4;
    int k  = chunk * 16 + row;
    int c0 = s * 32 + khalf * 16 + 2 * v;
    float f0 = -2.0f * E[k * C_DIM + c0];
    float f1 = -2.0f * E[k * C_DIM + c0 + 1];
    cbw[id] = pack_bf16(f0, f1);
}

// ---- Main kernel: WMMA distance GEMM + running argmin + gather + loss ------
// Each wave owns 32 pixels = two 16-row A tiles sharing every B tile:
// 4 WMMA per 4 global_load_b128, two independent accumulator chains.
// The per-code norm ‖e‖^2 rides in the WMMA C operand (D = A*B + C), so the
// full distance appears directly in the accumulator output.
__global__ __launch_bounds__(256) void vq_main(const float* __restrict__ z,
                                               const float* __restrict__ E,
                                               const float* __restrict__ norms,
                                               const uint32_t* __restrict__ cb,
                                               float* __restrict__ outq,
                                               float* __restrict__ outidx,
                                               float* __restrict__ partials) {
    __shared__ float norms_lds[K_CODES];
    __shared__ int   idx_lds[WAVES_PER_BLOCK][32];

    const int tid  = threadIdx.x;
    const int lane = tid & 31;
    const int wave = tid >> 5;

    for (int i = tid; i < K_CODES; i += 256) norms_lds[i] = norms[i];
    __syncthreads();

    const int pix_base = blockIdx.x * PIX_PER_BLOCK + wave * PIX_PER_WAVE;
    const int row   = lane & 15;     // pixel row M within tile / code column N
    const int khalf = lane >> 4;     // which K half this lane carries
    const int b  = pix_base >> 12;   // all 32 pixels share the image (4096 | 32)
    const int hw0 = (pix_base + row)      & (IMG_HW - 1);
    const int hw1 = (pix_base + 16 + row) & (IMG_HW - 1);
    const float* zp0 = z + (size_t)b * IMG_CHW + hw0;  // + c*IMG_HW per channel
    const float* zp1 = z + (size_t)b * IMG_CHW + hw1;

    // A tiles (2 x 16 pixels x 64 channels) in WMMA A layout, bf16.
    // A VGPR v (kstep s): channel = s*32 + 16*khalf + 2*v + {0,1}
    v16bf a0, a1, a2, a3;
    #pragma unroll
    for (int v = 0; v < 8; ++v) {
        int cb0 = 16 * khalf + 2 * v;
        a0[2 * v]     = (bf16_t)zp0[(size_t)(cb0 + 0) * IMG_HW];
        a0[2 * v + 1] = (bf16_t)zp0[(size_t)(cb0 + 1) * IMG_HW];
        a1[2 * v]     = (bf16_t)zp0[(size_t)(32 + cb0 + 0) * IMG_HW];
        a1[2 * v + 1] = (bf16_t)zp0[(size_t)(32 + cb0 + 1) * IMG_HW];
        a2[2 * v]     = (bf16_t)zp1[(size_t)(cb0 + 0) * IMG_HW];
        a2[2 * v + 1] = (bf16_t)zp1[(size_t)(cb0 + 1) * IMG_HW];
        a3[2 * v]     = (bf16_t)zp1[(size_t)(32 + cb0 + 0) * IMG_HW];
        a3[2 * v + 1] = (bf16_t)zp1[(size_t)(32 + cb0 + 1) * IMG_HW];
    }

    float bestv0[8], bestv1[8];
    int   besti0[8], besti1[8];
    #pragma unroll
    for (int r = 0; r < 8; ++r) {
        bestv0[r] = 3.4e38f; besti0[r] = 0;
        bestv1[r] = 3.4e38f; besti1[r] = 0;
    }

    for (int chunk = 0; chunk < N_CHUNKS; ++chunk) {
        const uint32_t* c0p = cb + (size_t)chunk * 512 + lane * 8;
        const uint32_t* c1p = c0p + 256;
        union { uint32_t u[8]; v16bf v; } B0, B1;
        *(uint4*)&B0.u[0] = *(const uint4*)&c0p[0];
        *(uint4*)&B0.u[4] = *(const uint4*)&c0p[4];
        *(uint4*)&B1.u[0] = *(const uint4*)&c1p[0];
        *(uint4*)&B1.u[4] = *(const uint4*)&c1p[4];
        __builtin_prefetch(c0p + 512, 0, 1);   // speculative; safe past the end

        // C operand = splat(‖e‖^2) for this lane's code column; shared by both
        // chains, so D = ‖e‖^2 - 2 z.e lands directly in the accumulators.
        const int   code = chunk * 16 + row;
        const float nrm  = norms_lds[code];
        v8f cinit;
        #pragma unroll
        for (int r = 0; r < 8; ++r) cinit[r] = nrm;

        v8f accA, accB;
        accA = __builtin_amdgcn_wmma_f32_16x16x32_bf16(false, a0, false, B0.v,
                                                       (short)0, cinit, false, false);
        accB = __builtin_amdgcn_wmma_f32_16x16x32_bf16(false, a2, false, B0.v,
                                                       (short)0, cinit, false, false);
        accA = __builtin_amdgcn_wmma_f32_16x16x32_bf16(false, a1, false, B1.v,
                                                       (short)0, accA, false, false);
        accB = __builtin_amdgcn_wmma_f32_16x16x32_bf16(false, a3, false, B1.v,
                                                       (short)0, accB, false, false);

        #pragma unroll
        for (int r = 0; r < 8; ++r) {
            float dA = accA[r];
            bool bA = dA < bestv0[r];
            bestv0[r] = bA ? dA   : bestv0[r];
            besti0[r] = bA ? code : besti0[r];
            float dB = accB[r];
            bool bB = dB < bestv1[r];
            bestv1[r] = bB ? dB   : bestv1[r];
            besti1[r] = bB ? code : besti1[r];
        }
    }

    // Reduce argmin across the 16 code columns (xor masks stay inside halves).
    #pragma unroll
    for (int r = 0; r < 8; ++r) {
        #pragma unroll
        for (int off = 8; off >= 1; off >>= 1) {
            float ov = __shfl_xor(bestv0[r], off, 32);
            int   oi = __shfl_xor(besti0[r], off, 32);
            if (ov < bestv0[r] || (ov == bestv0[r] && oi < besti0[r])) {
                bestv0[r] = ov; besti0[r] = oi;
            }
            float pv = __shfl_xor(bestv1[r], off, 32);
            int   pi = __shfl_xor(besti1[r], off, 32);
            if (pv < bestv1[r] || (pv == bestv1[r] && pi < besti1[r])) {
                bestv1[r] = pv; besti1[r] = pi;
            }
        }
    }
    // Lane 0 holds rows 0-7, lane 16 holds rows 8-15 (C/D layout).
    if (row == 0) {
        #pragma unroll
        for (int r = 0; r < 8; ++r) {
            int m = khalf * 8 + r;
            idx_lds[wave][m]          = besti0[r];
            idx_lds[wave][16 + m]     = besti1[r];
            outidx[pix_base + m]      = (float)besti0[r];
            outidx[pix_base + 16 + m] = (float)besti1[r];
        }
    }
    __syncthreads();

    // Gather code rows (exact fp32), write quantized output, accumulate loss.
    float lsum = 0.f;
    #pragma unroll
    for (int t = 0; t < 2; ++t) {
        const float* zrd = t ? zp1 : zp0;
        float* qwr = outq + (zrd - z);
        const int idx = idx_lds[wave][t * 16 + row];
        const float* Erow = E + (size_t)idx * C_DIM + khalf * 32;
        #pragma unroll 8
        for (int j = 0; j < 32; ++j) {
            int c = khalf * 32 + j;
            float q  = Erow[j];
            float zv = zrd[(size_t)c * IMG_HW];
            qwr[(size_t)c * IMG_HW] = q;
            float dd = q - zv;
            lsum += dd * dd;
        }
    }
    #pragma unroll
    for (int off = 16; off >= 1; off >>= 1) lsum += __shfl_xor(lsum, off, 32);
    if (lane == 0) partials[blockIdx.x * WAVES_PER_BLOCK + wave] = lsum;
}

// ---- Deterministic loss reduction ------------------------------------------
__global__ __launch_bounds__(256) void vq_loss(const float* __restrict__ partials,
                                               float* __restrict__ loss_out) {
    __shared__ float red[256];
    float s = 0.f;
    for (int i = threadIdx.x; i < N_WAVES; i += 256) s += partials[i];
    red[threadIdx.x] = s;
    __syncthreads();
    for (int w = 128; w > 0; w >>= 1) {
        if (threadIdx.x < w) red[threadIdx.x] += red[threadIdx.x + w];
        __syncthreads();
    }
    if (threadIdx.x == 0)
        loss_out[0] = 0.25f * red[0] / (float)(N_PIX * C_DIM);
}

extern "C" void kernel_launch(void* const* d_in, const int* in_sizes, int n_in,
                              void* d_out, int out_size, void* d_ws, size_t ws_size,
                              hipStream_t stream) {
    const float* z = (const float*)d_in[0];        // (32, 64, 64, 64) fp32
    const float* E = (const float*)d_in[1];        // (2048, 64) fp32

    float*    norms    = (float*)d_ws;                               // 2048 f
    uint32_t* cbw      = (uint32_t*)(norms + K_CODES);               // 65536 dw
    float*    partials = (float*)(cbw + CB_DWORDS);                  // 4096 f

    float* outq   = (float*)d_out;                  // (32,64,64,64) quantized
    float* outidx = outq + (size_t)N_PIX * C_DIM;   // (32,64,64) indices
    float* loss   = outidx + N_PIX;                 // scalar

    vq_norms     <<<(K_CODES + 255) / 256, 256, 0, stream>>>(E, norms);
    vq_swizzle_cb<<<CB_DWORDS / 256,       256, 0, stream>>>(E, cbw);
    vq_main      <<<N_BLOCKS,              256, 0, stream>>>(z, E, norms, cbw,
                                                             outq, outidx, partials);
    vq_loss      <<<1,                     256, 0, stream>>>(partials, loss);
}
